// KANConv2d_57286273794765
// MI455X (gfx1250) — compile-verified, matrix operand order checked by reading, not run on hardware
//
#include <hip/hip_runtime.h>
#include <hip/hip_bf16.h>

typedef __attribute__((ext_vector_type(16))) __bf16 v16bf;
typedef __attribute__((ext_vector_type(8)))  __bf16 v8bf;
typedef __attribute__((ext_vector_type(8)))  float  v8f;
typedef __attribute__((ext_vector_type(4)))  float  v4f;

#define B_   16
#define C_   64
#define H_   64
#define W_   64
#define O_   128
#define HP   66
#define WP   66
#define SB   7            // live basis funcs; s=7..9 are identically zero
#define E_   (C_ + SB*C_) // 512 effective fused input channels
#define KTAPS 9

__device__ __forceinline__ float knot(int i) { return 0.5f * (float)i - 2.5f; }

// ---------------------------------------------------------------------------
// Phase 1a: F[b][y][x][e] bf16, NHWC over the padded 66x66 grid.
//   e in [0,64)    : silu(xpad)
//   e in [64,512)  : cubic B-spline basis s=0..6 of xpad (uniform knots)
// Border positions use xpad = 0 -> silu(0)=0 but basis(0) != 0 (matches ref).
// ---------------------------------------------------------------------------
__global__ void kan_build_feat(const float* __restrict__ x,
                               __bf16* __restrict__ F) {
    int id = blockIdx.x * blockDim.x + threadIdx.x;
    const int total = B_ * HP * WP * C_;
    if (id >= total) return;

    const int c  = id & (C_ - 1);
    int pos      = id >> 6;            // / C_
    const int xc = pos % WP;  pos /= WP;
    const int yc = pos % HP;
    const int b  = pos / HP;

    float v = 0.0f;
    if (yc >= 1 && yc <= H_ && xc >= 1 && xc <= W_) {
        v = __builtin_nontemporal_load(
                x + (((size_t)b * C_ + c) * H_ + (yc - 1)) * W_ + (xc - 1));
    }

    // silu
    const float sv = v * (1.0f / (1.0f + expf(-v)));

    // de Boor, uniform knots t(i) = 0.5*i - 2.5, fully unrolled
    float bb[10];
    #pragma unroll
    for (int i = 0; i < 10; ++i)
        bb[i] = (v >= knot(i) && v < knot(i + 1)) ? 1.0f : 0.0f;
    #pragma unroll
    for (int d = 1; d <= 3; ++d) {
        const int n = 10 - d;
        #pragma unroll
        for (int i = 0; i < 9; ++i) {
            if (i < n) {
                float left  = (v - knot(i))         / (knot(i + d)     - knot(i))     * bb[i];
                float right = (knot(i + d + 1) - v) / (knot(i + d + 1) - knot(i + 1)) * bb[i + 1];
                bb[i] = left + right;
            }
        }
    }

    __bf16* Fp = F + ((size_t)(b * HP + yc) * WP + xc) * E_;
    Fp[c] = (__bf16)sv;
    #pragma unroll
    for (int s = 0; s < SB; ++s)
        Fp[C_ + s * C_ + c] = (__bf16)bb[s];
}

// ---------------------------------------------------------------------------
// Phase 1b: Wt[tap][o][e] bf16 — fused conv weights, K contiguous per N row.
//   e in [0,64)   : beta[o][c][kh][kw]
//   e in [64,512) : spline[o][c][kh][kw] * coeffs[s][o][c][kh][kw]
// ---------------------------------------------------------------------------
__global__ void kan_build_wt(const float* __restrict__ beta,
                             const float* __restrict__ spline,
                             const float* __restrict__ coeffs,
                             __bf16* __restrict__ Wt) {
    int id = blockIdx.x * blockDim.x + threadIdx.x;
    const int total = O_ * C_ * KTAPS;
    if (id >= total) return;

    const int tap = id % KTAPS;
    int rest      = id / KTAPS;
    const int c   = rest & (C_ - 1);
    const int o   = rest >> 6;
    const int kh  = tap / 3, kw = tap % 3;

    const size_t wIdx = (((size_t)o * C_ + c) * 3 + kh) * 3 + kw;
    __bf16* Wp = Wt + ((size_t)tap * O_ + o) * E_;

    Wp[c] = (__bf16)beta[wIdx];
    const float sw = spline[wIdx];
    #pragma unroll
    for (int s = 0; s < SB; ++s) {
        const size_t cIdx = ((((size_t)s * O_ + o) * C_ + c) * 3 + kh) * 3 + kw;
        Wp[C_ + s * C_ + c] = (__bf16)(sw * coeffs[cIdx]);
    }
}

// ---------------------------------------------------------------------------
// Helpers: load a 16x32 bf16 fragment (ISA A/B layout) from a K-contiguous row
// ---------------------------------------------------------------------------
__device__ __forceinline__ v16bf load_frag(const __bf16* __restrict__ row, int e0) {
    v8bf lo = *(const v8bf*)(row + e0);
    v8bf hi = *(const v8bf*)(row + e0 + 16);
    v16bf f;
    #pragma unroll
    for (int i = 0; i < 8; ++i) { f[i] = lo[i]; f[i + 8] = hi[i]; }
    return f;
}

// ---------------------------------------------------------------------------
// Phase 2: implicit GEMM with 2x2 register tiling.
// Block = one output row (b,h): 64 pixels x 128 out-channels, 256 thr = 8 waves.
// Wave (mrow = wv>>2, ncol = wv&3) owns a 32-pixel x 32-channel macro-tile:
//   per K-slice: 2 A frags + 2 B frags -> 4 v_wmma_f32_16x16x32_bf16
//   (2 b128 loads per WMMA instead of 4 -> 2x arithmetic intensity).
// K loop: 9 taps x 16 slices = 144 slices, 576 WMMAs per wave.
// ---------------------------------------------------------------------------
__global__ void __launch_bounds__(256)
kan_wmma_gemm(const __bf16* __restrict__ F,
              const __bf16* __restrict__ Wt,
              float* __restrict__ out) {
    const int tid  = threadIdx.x;
    const int wv   = tid >> 5;
    const int lane = tid & 31;
    const int nl   = lane & 15;   // A: M row index; B: N column index
    const int hlf  = lane >> 4;   // selects K chunks per ISA A/B layout

    const int b  = blockIdx.x >> 6;       // 1024 blocks = 16 b x 64 h
    const int h  = blockIdx.x & 63;
    const int m0 = (wv >> 2) << 5;        // pixel-w base of this wave (0 or 32)
    const int n0 = (wv & 3) << 5;         // out-channel base (0,32,64,96)

    v8f acc00 = {}, acc01 = {}, acc10 = {}, acc11 = {};

    #pragma unroll
    for (int kh = 0; kh < 3; ++kh) {
        #pragma unroll
        for (int kw = 0; kw < 3; ++kw) {
            const __bf16* Arow0 =
                F + ((size_t)(b * HP + (h + kh)) * WP + (m0 + nl + kw)) * E_ + hlf * 8;
            const __bf16* Arow1 = Arow0 + (size_t)16 * E_;
            const __bf16* Brow0 =
                Wt + ((size_t)(kh * 3 + kw) * O_ + (n0 + nl)) * E_ + hlf * 8;
            const __bf16* Brow1 = Brow0 + (size_t)16 * E_;
            #pragma unroll 2
            for (int e0 = 0; e0 < E_; e0 += 32) {
                v16bf a0 = load_frag(Arow0, e0);
                v16bf a1 = load_frag(Arow1, e0);
                v16bf b0 = load_frag(Brow0, e0);
                v16bf b1 = load_frag(Brow1, e0);
                acc00 = __builtin_amdgcn_wmma_f32_16x16x32_bf16(
                            false, a0, false, b0, (short)0, acc00, false, false);
                acc01 = __builtin_amdgcn_wmma_f32_16x16x32_bf16(
                            false, a0, false, b1, (short)0, acc01, false, false);
                acc10 = __builtin_amdgcn_wmma_f32_16x16x32_bf16(
                            false, a1, false, b0, (short)0, acc10, false, false);
                acc11 = __builtin_amdgcn_wmma_f32_16x16x32_bf16(
                            false, a1, false, b1, (short)0, acc11, false, false);
            }
        }
    }

    // D layout: VGPR r, lane-half hlf -> M = hlf*8 + r (consecutive w), N = nl.
    // Output is write-once -> non-temporal stores keep F resident in L2.
    const size_t rowbase = ((size_t)b * O_ * H_ + h) * W_;
    #pragma unroll
    for (int i = 0; i < 2; ++i) {
        #pragma unroll
        for (int j = 0; j < 2; ++j) {
            const v8f* accp = (i == 0) ? ((j == 0) ? &acc00 : &acc01)
                                       : ((j == 0) ? &acc10 : &acc11);
            const v8f a = *accp;
            float* po = out + rowbase + (size_t)(n0 + j * 16 + nl) * (H_ * W_)
                            + m0 + i * 16 + hlf * 8;
            v4f lo = { a[0], a[1], a[2], a[3] };
            v4f hi = { a[4], a[5], a[6], a[7] };
            __builtin_nontemporal_store(lo, (v4f*)po);
            __builtin_nontemporal_store(hi, (v4f*)(po + 4));
        }
    }
}

// ---------------------------------------------------------------------------
extern "C" void kernel_launch(void* const* d_in, const int* in_sizes, int n_in,
                              void* d_out, int out_size, void* d_ws, size_t ws_size,
                              hipStream_t stream) {
    (void)in_sizes; (void)n_in; (void)out_size; (void)ws_size;

    const float* x      = (const float*)d_in[0];
    const float* beta   = (const float*)d_in[1];
    const float* spline = (const float*)d_in[2];
    const float* coeffs = (const float*)d_in[3];
    float* out          = (float*)d_out;

    __bf16* F = (__bf16*)d_ws;                               // 71.4 MB (L2-resident)
    const size_t F_elems = (size_t)B_ * HP * WP * E_;
    __bf16* Wt = F + F_elems;                                // +1.2 MB

    {
        const int total = B_ * HP * WP * C_;                 // 4,460,544
        kan_build_feat<<<(total + 255) / 256, 256, 0, stream>>>(x, F);
    }
    {
        const int total = O_ * C_ * KTAPS;                   // 73,728
        kan_build_wt<<<(total + 255) / 256, 256, 0, stream>>>(beta, spline, coeffs, Wt);
    }
    {
        const int blocks = B_ * H_;                          // 1024 blocks x 8 waves
        kan_wmma_gemm<<<blocks, 256, 0, stream>>>(F, Wt, out);
    }
}
